// MSNATLayer_74088185856246
// MI455X (gfx1250) — matrix operand order repaired and checked by reading, not compiled, not run on hardware
//
#include <hip/hip_runtime.h>
#include <hip/hip_bf16.h>
#include <math.h>

typedef __attribute__((ext_vector_type(16))) _Float16 v16h;
typedef __attribute__((ext_vector_type(8)))  _Float16 v8h;
typedef __attribute__((ext_vector_type(8)))  float    v8f;

#define HEADS 4
#define CCH   128
#define HDIM  32

// ---------------------------------------------------------------------------
// Async global->LDS copy (gfx1250 GLOBAL_LOAD_ASYNC_TO_LDS_B128) via inline
// asm (the clang builtin's pointer address-space typing is not portable).
// LDS destination = low 32 bits of the generic pointer (ISA 10.2).
// Tracked with ASYNCcnt; waited with s_wait_asynccnt.
// ---------------------------------------------------------------------------
#if defined(__HIP_DEVICE_COMPILE__) && defined(__gfx1250__)
#define USE_ASYNC_LDS 1
#else
#define USE_ASYNC_LDS 0
#endif

__device__ __forceinline__ void copy16_g2l(_Float16* dst, const _Float16* src) {
#if USE_ASYNC_LDS
    unsigned lds_addr = (unsigned)(unsigned long long)dst;
    asm volatile("global_load_async_to_lds_b128 %0, %1, off"
                 :
                 : "v"(lds_addr), "v"(src)
                 : "memory");
#else
    *(v8h*)dst = *(const v8h*)src;
#endif
}

__device__ __forceinline__ void wait_async_copies() {
#if USE_ASYNC_LDS
#if __has_builtin(__builtin_amdgcn_s_wait_asynccnt)
    __builtin_amdgcn_s_wait_asynccnt(0);
#else
    asm volatile("s_wait_asynccnt 0x0" ::: "memory");
#endif
#endif
}

// ---------------------------------------------------------------------------
// f32 -> f16 converter (for weights)
// ---------------------------------------------------------------------------
__global__ void f32_to_f16_kernel(const float* __restrict__ in,
                                  _Float16* __restrict__ out, int n) {
    int i = blockIdx.x * blockDim.x + threadIdx.x;
    if (i < n) out[i] = (_Float16)in[i];
}

// ---------------------------------------------------------------------------
// LayerNorm over channels, input NCHW f32 -> output [P, C] f16 (pixel-major)
// ---------------------------------------------------------------------------
__global__ void ln_nchw_kernel(const float* __restrict__ x,
                               const float* __restrict__ g,
                               const float* __restrict__ b,
                               _Float16* __restrict__ out,
                               int C, int HW, int P) {
    int p = blockIdx.x * blockDim.x + threadIdx.x;
    if (p >= P) return;
    int bt = p / HW, hw = p % HW;
    const float* xp = x + (size_t)bt * C * HW + hw;
    float s = 0.f, s2 = 0.f;
    for (int c = 0; c < C; ++c) {
        float v = xp[(size_t)c * HW];
        s += v; s2 += v * v;
    }
    float mean = s / C;
    float var  = s2 / C - mean * mean;
    float r    = rsqrtf(var + 1e-5f);
    _Float16* op = out + (size_t)p * C;
    for (int c = 0; c < C; ++c) {
        float v = xp[(size_t)c * HW];
        op[c] = (_Float16)((v - mean) * r * g[c] + b[c]);
    }
}

// LayerNorm over row-major [P, C] f32 -> [P, C] f16
__global__ void ln_row_kernel(const float* __restrict__ x,
                              const float* __restrict__ g,
                              const float* __restrict__ b,
                              _Float16* __restrict__ out,
                              int C, int P) {
    int p = blockIdx.x * blockDim.x + threadIdx.x;
    if (p >= P) return;
    const float* xp = x + (size_t)p * C;
    float s = 0.f, s2 = 0.f;
    for (int c = 0; c < C; ++c) { float v = xp[c]; s += v; s2 += v * v; }
    float mean = s / C;
    float var  = s2 / C - mean * mean;
    float r    = rsqrtf(var + 1e-5f);
    _Float16* op = out + (size_t)p * C;
    for (int c = 0; c < C; ++c)
        op[c] = (_Float16)((xp[c] - mean) * r * g[c] + b[c]);
}

// ---------------------------------------------------------------------------
// WMMA GEMM:  out[M,N] = A[M,K](f16) * W[N,K]^T(f16) + bias
// Block: 256 thr = 8 waves; block tile 128(M) x 128(N); wave tile 32x64
// (2x4 WMMA 16x16x32 tiles -> 8 WMMA per K-step). Double-buffered LDS with
// async global->LDS staging pipelined against the WMMAs.
// ACT: 0 none, 1 exact GELU. WF/WH: write f32 / f16 outputs.
// M % 128 == 0, N % 128 == 0, K % 32 == 0 (guaranteed by caller).
// ---------------------------------------------------------------------------
template <int ACT, bool WF, bool WH>
__global__ void __launch_bounds__(256)
gemm_wmma_kernel(const _Float16* __restrict__ A,
                 const _Float16* __restrict__ Wt,   // [N,K] row-major
                 const float* __restrict__ bias,
                 float* __restrict__ outF,
                 _Float16* __restrict__ outH,
                 int M, int N, int K) {
    __shared__ _Float16 As[2][128 * 40];   // 32 halves + 8 pad per row
    __shared__ _Float16 Bs[2][128 * 40];

    const int tid  = threadIdx.x;
    const int lane = tid & 31;
    const int l15  = lane & 15;
    const int lhi  = lane >> 4;
    const int wave = tid >> 5;          // 0..7
    const int waveM = wave & 3;         // 0..3 -> M offset *32
    const int waveN = wave >> 2;        // 0..1 -> N offset *64

    const int m0 = blockIdx.x * 128;
    const int n0 = blockIdx.y * 128;

    // stage one 128x32-half tile of A and of B into LDS buffer `buf`
    auto stage = [&](int buf, int k0) {
#pragma unroll
        for (int u = 0; u < 2; ++u) {
            int ch  = tid * 2 + u;        // 512 chunks of 16B per tile
            int row = ch >> 2;
            int off = (ch & 3) * 8;
            copy16_g2l(&As[buf][row * 40 + off],
                       A + (size_t)(m0 + row) * K + k0 + off);
            copy16_g2l(&Bs[buf][row * 40 + off],
                       Wt + (size_t)(n0 + row) * K + k0 + off);
        }
    };

    v8f c[2][4] = {};

    const int nk = K >> 5;
    stage(0, 0);
    for (int kt = 0; kt < nk; ++kt) {
        const int buf = kt & 1;
        wait_async_copies();
        __syncthreads();                 // tile `buf` visible to all waves
        if (kt + 1 < nk)
            stage(buf ^ 1, (kt + 1) << 5);   // overlap next stage with math

        // A fragments (ISA 16-bit 16x32 layout: K = h + 8*((h>>3)+(lane>>4)))
        v16h af[2];
#pragma unroll
        for (int s = 0; s < 2; ++s) {
            int row = waveM * 32 + s * 16 + l15;
            v8h lo = *(const v8h*)&As[buf][row * 40 + 8 * lhi];
            v8h hi = *(const v8h*)&As[buf][row * 40 + 16 + 8 * lhi];
#pragma unroll
            for (int h = 0; h < 8; ++h) { af[s][h] = lo[h]; af[s][h + 8] = hi[h]; }
        }
        // B fragments (lane = N, K = h + 8*((h>>3)+(lane>>4)))
        v16h bf[4];
#pragma unroll
        for (int s = 0; s < 4; ++s) {
            int col = waveN * 64 + s * 16 + l15;
            v8h lo = *(const v8h*)&Bs[buf][col * 40 + 8 * lhi];
            v8h hi = *(const v8h*)&Bs[buf][col * 40 + 16 + 8 * lhi];
#pragma unroll
            for (int h = 0; h < 8; ++h) { bf[s][h] = lo[h]; bf[s][h + 8] = hi[h]; }
        }

#pragma unroll
        for (int i = 0; i < 2; ++i)
#pragma unroll
            for (int j = 0; j < 4; ++j)
                c[i][j] = __builtin_amdgcn_wmma_f32_16x16x32_f16(
                    false, af[i], false, bf[j], (short)0, c[i][j], false, false);
    }

    // epilogue: C layout -> M = r + 8*(lane>>4), N = lane&15 per 16x16 tile
#pragma unroll
    for (int i = 0; i < 2; ++i) {
#pragma unroll
        for (int j = 0; j < 4; ++j) {
            int gn = n0 + waveN * 64 + j * 16 + l15;
            float bv = bias[gn];
#pragma unroll
            for (int r = 0; r < 8; ++r) {
                int gm = m0 + waveM * 32 + i * 16 + r + 8 * lhi;
                float v = c[i][j][r] + bv;
                if constexpr (ACT == 1)
                    v = 0.5f * v * (1.0f + erff(v * 0.70710678118654752f));
                size_t o = (size_t)gm * N + gn;
                if constexpr (WF) outF[o] = v;
                if constexpr (WH) outH[o] = (_Float16)v;
            }
        }
    }
}

// ---------------------------------------------------------------------------
// Neighborhood attention core. qkv: [P, 3*C] f16, packed [3][heads][hd].
// One thread per (pixel, head); online softmax over WxW window.
// ---------------------------------------------------------------------------
template <int WIN>
__global__ void nat_kernel(const _Float16* __restrict__ qkv,
                           const float* __restrict__ rpb,
                           _Float16* __restrict__ out,
                           int B, int H, int Wd) {
    int t = blockIdx.x * blockDim.x + threadIdx.x;
    int HW = H * Wd;
    int total = B * HW * HEADS;
    if (t >= total) return;
    int n = t & (HEADS - 1);
    int p = t >> 2;                      // HEADS == 4
    int b  = p / HW;
    int rem = p % HW;
    int i = rem / Wd, j = rem % Wd;

    const float scale = 0.17677669529663687f;   // 32^-0.5
    float q[HDIM];
    const _Float16* qp = qkv + (size_t)p * (3 * CCH) + n * HDIM;
#pragma unroll
    for (int d = 0; d < HDIM; ++d) q[d] = (float)qp[d] * scale;

    int si = i - WIN / 2; if (si < 0) si = 0; if (si > H - WIN) si = H - WIN;
    int sj = j - WIN / 2; if (sj < 0) sj = 0; if (sj > Wd - WIN) sj = Wd - WIN;

    float m = -1e30f, ssum = 0.f;
    float acc[HDIM];
#pragma unroll
    for (int d = 0; d < HDIM; ++d) acc[d] = 0.f;

    const float* rp = rpb + (size_t)n * (2 * WIN - 1) * (2 * WIN - 1);

    for (int a = 0; a < WIN; ++a) {
        int ki = si + a;
        int bi = (WIN - 1) + a - (i - si);
        for (int c2 = 0; c2 < WIN; ++c2) {
            int kj = sj + c2;
            int bj = (WIN - 1) + c2 - (j - sj);
            size_t pn = (size_t)(b * H + ki) * Wd + kj;
            const _Float16* kp = qkv + pn * (3 * CCH) + CCH + n * HDIM;
            float lg = 0.f;
#pragma unroll
            for (int d = 0; d < HDIM; ++d) lg += q[d] * (float)kp[d];
            lg += rp[bi * (2 * WIN - 1) + bj];

            float mnew = fmaxf(m, lg);
            float corr = __expf(m - mnew);
            float w    = __expf(lg - mnew);
            ssum = ssum * corr + w;
            const _Float16* vp = kp + CCH;
#pragma unroll
            for (int d = 0; d < HDIM; ++d)
                acc[d] = acc[d] * corr + w * (float)vp[d];
            m = mnew;
        }
    }
    float inv = 1.f / ssum;
    _Float16* op = out + (size_t)p * CCH + n * HDIM;
#pragma unroll
    for (int d = 0; d < HDIM; ++d) op[d] = (_Float16)(acc[d] * inv);
}

// ---------------------------------------------------------------------------
// x2[p,c] = x_nchw[b,c,hw] + a1[p,c] + a2[p,c]
// ---------------------------------------------------------------------------
__global__ void residual_kernel(const float* __restrict__ x,
                                const float* __restrict__ a1,
                                const float* __restrict__ a2,
                                float* __restrict__ x2,
                                int C, int HW, int total) {
    int idx = blockIdx.x * blockDim.x + threadIdx.x;
    if (idx >= total) return;
    int p = idx / C, c = idx % C;
    int b = p / HW, hw = p % HW;
    x2[idx] = x[(size_t)b * C * HW + (size_t)c * HW + hw] + a1[idx] + a2[idx];
}

// out_nchw = x2[p,c] + y[p,c]
__global__ void final_kernel(const float* __restrict__ x2,
                             const float* __restrict__ y,
                             float* __restrict__ out,
                             int C, int HW, int total) {
    int idx = blockIdx.x * blockDim.x + threadIdx.x;
    if (idx >= total) return;
    int b = idx / (C * HW);
    int r = idx % (C * HW);
    int c = r / HW, hw = r % HW;
    size_t pc = (size_t)(b * HW + hw) * C + c;
    out[idx] = x2[pc] + y[pc];
}

// ---------------------------------------------------------------------------
extern "C" void kernel_launch(void* const* d_in, const int* in_sizes, int n_in,
                              void* d_out, int out_size, void* d_ws, size_t ws_size,
                              hipStream_t stream) {
    (void)in_sizes; (void)n_in; (void)out_size; (void)ws_size;
    const int B = 2, C = 128, H = 56, Wd = 56;
    const int HW = H * Wd, P = B * HW;

    const float* x       = (const float*)d_in[0];
    const float* ln1_g   = (const float*)d_in[1];
    const float* ln1_b   = (const float*)d_in[2];
    const float* qkv_w1  = (const float*)d_in[3];
    const float* qkv_b1  = (const float*)d_in[4];
    const float* proj_w1 = (const float*)d_in[5];
    const float* proj_b1 = (const float*)d_in[6];
    const float* rpb1    = (const float*)d_in[7];
    const float* qkv_w2  = (const float*)d_in[8];
    const float* qkv_b2  = (const float*)d_in[9];
    const float* proj_w2 = (const float*)d_in[10];
    const float* proj_b2 = (const float*)d_in[11];
    const float* rpb2    = (const float*)d_in[12];
    const float* ln2_g   = (const float*)d_in[13];
    const float* ln2_b   = (const float*)d_in[14];
    const float* fc1_w   = (const float*)d_in[15];
    const float* fc1_b   = (const float*)d_in[16];
    const float* fc2_w   = (const float*)d_in[17];
    const float* fc2_b   = (const float*)d_in[18];
    float* out = (float*)d_out;

    char* wp = (char*)d_ws;
    auto alloc = [&](size_t bytes) -> char* {
        char* r = wp; wp += (bytes + 255) & ~(size_t)255; return r;
    };
    _Float16* xn16   = (_Float16*)alloc((size_t)P * C * 2);
    _Float16* qkvw1h = (_Float16*)alloc((size_t)384 * 128 * 2);
    _Float16* qkvw2h = (_Float16*)alloc((size_t)384 * 128 * 2);
    _Float16* pjw1h  = (_Float16*)alloc((size_t)128 * 128 * 2);
    _Float16* pjw2h  = (_Float16*)alloc((size_t)128 * 128 * 2);
    _Float16* fc1wh  = (_Float16*)alloc((size_t)256 * 128 * 2);
    _Float16* fc2wh  = (_Float16*)alloc((size_t)128 * 256 * 2);
    _Float16* qkv1h  = (_Float16*)alloc((size_t)P * 384 * 2);
    _Float16* qkv2h  = (_Float16*)alloc((size_t)P * 384 * 2);
    _Float16* ao1h   = (_Float16*)alloc((size_t)P * C * 2);
    _Float16* ao2h   = (_Float16*)alloc((size_t)P * C * 2);
    float*    a1f    = (float*)alloc((size_t)P * C * 4);
    float*    a2f    = (float*)alloc((size_t)P * C * 4);
    float*    x2f    = (float*)alloc((size_t)P * C * 4);
    _Float16* xn2h   = (_Float16*)alloc((size_t)P * C * 2);
    _Float16* hh     = (_Float16*)alloc((size_t)P * 256 * 2);
    float*    yf     = (float*)alloc((size_t)P * C * 4);

    // 0) weight conversions
    auto cvt = [&](const float* src, _Float16* dst, int n) {
        f32_to_f16_kernel<<<(n + 255) / 256, 256, 0, stream>>>(src, dst, n);
    };
    cvt(qkv_w1, qkvw1h, 384 * 128);
    cvt(qkv_w2, qkvw2h, 384 * 128);
    cvt(proj_w1, pjw1h, 128 * 128);
    cvt(proj_w2, pjw2h, 128 * 128);
    cvt(fc1_w, fc1wh, 256 * 128);
    cvt(fc2_w, fc2wh, 128 * 256);

    // 1) LN1 (NCHW -> [P,C] f16)
    ln_nchw_kernel<<<(P + 255) / 256, 256, 0, stream>>>(x, ln1_g, ln1_b, xn16, C, HW, P);

    // 2) QKV GEMMs: [P,128] x [384,128]^T -> [P,384] f16
    gemm_wmma_kernel<0, false, true><<<dim3(P / 128, 384 / 128), 256, 0, stream>>>(
        xn16, qkvw1h, qkv_b1, nullptr, qkv1h, P, 384, 128);
    gemm_wmma_kernel<0, false, true><<<dim3(P / 128, 384 / 128), 256, 0, stream>>>(
        xn16, qkvw2h, qkv_b2, nullptr, qkv2h, P, 384, 128);

    // 3) neighborhood attention cores
    {
        int total = P * HEADS;
        nat_kernel<7><<<(total + 255) / 256, 256, 0, stream>>>(qkv1h, rpb1, ao1h, B, H, Wd);
        nat_kernel<5><<<(total + 255) / 256, 256, 0, stream>>>(qkv2h, rpb2, ao2h, B, H, Wd);
    }

    // 4) projections -> f32
    gemm_wmma_kernel<0, true, false><<<dim3(P / 128, 128 / 128), 256, 0, stream>>>(
        ao1h, pjw1h, proj_b1, a1f, nullptr, P, 128, 128);
    gemm_wmma_kernel<0, true, false><<<dim3(P / 128, 128 / 128), 256, 0, stream>>>(
        ao2h, pjw2h, proj_b2, a2f, nullptr, P, 128, 128);

    // 5) residual: x2 = x + a1 + a2   (pixel-major f32)
    residual_kernel<<<(P * C + 255) / 256, 256, 0, stream>>>(x, a1f, a2f, x2f, C, HW, P * C);

    // 6) LN2 ([P,C] f32 -> f16)
    ln_row_kernel<<<(P + 255) / 256, 256, 0, stream>>>(x2f, ln2_g, ln2_b, xn2h, C, P);

    // 7) MLP: fc1 (+GELU, f16 out), fc2 (f32 out)
    gemm_wmma_kernel<1, false, true><<<dim3(P / 128, 256 / 128), 256, 0, stream>>>(
        xn2h, fc1wh, fc1_b, nullptr, hh, P, 256, 128);
    gemm_wmma_kernel<0, true, false><<<dim3(P / 128, 128 / 128), 256, 0, stream>>>(
        hh, fc2wh, fc2_b, yf, nullptr, P, 128, 256);

    // 8) final residual + transpose back to NCHW
    final_kernel<<<(B * C * HW + 255) / 256, 256, 0, stream>>>(
        x2f, yf, out, C, HW, B * C * HW);
}
